// STGumbel_AR_Tree_33079838114212
// MI455X (gfx1250) — matrix operand order repaired and checked by reading, not compiled, not run on hardware
//
#include <hip/hip_runtime.h>
#include <stdint.h>

typedef __attribute__((ext_vector_type(16))) _Float16 v16h;
typedef __attribute__((ext_vector_type(8)))  _Float16 v8h;
typedef __attribute__((ext_vector_type(8)))  float    v8f;
typedef __attribute__((ext_vector_type(4)))  unsigned int v4u;
typedef __attribute__((ext_vector_type(8)))  int      v8i;
typedef __attribute__((ext_vector_type(4)))  int      v4i;

#define BATCH 16
#define LSEQ  32
#define WORDD 512
#define HALFD 512
#define HDIM  1024
#define RHID  256
#define CDIM  3072   // 3*H
#define GCOMP 5120   // 5*H
#define MAXN  64
#define GSTRIDE (512 * WORDD)   // elements between gate blocks of Wih/Whh

__device__ __forceinline__ float sigf(float x) { return 1.0f / (1.0f + __expf(-x)); }

// A-operand pack, contiguous K order: lane<16 holds K{0..15},
// lane>=16 holds K{16..31}; caller passes lo = &row[k0 + (lane<16?0:16)].
__device__ __forceinline__ v16h pack_a(const _Float16* lo) {
    union { v16h v; v8h h[2]; } u;
    u.h[0] = *(const v8h*)(lo);
    u.h[1] = *(const v8h*)(lo + 8);
    return u.v;
}

__device__ __forceinline__ v8f wmma_f16(v16h a, v16h b, v8f c) {
    return __builtin_amdgcn_wmma_f32_16x16x32_f16(false, a, false, b, (short)0, c, false, false);
}

// ---------------------------------------------------------------- conversion
__global__ void cvt_f32_to_f16(const float* __restrict__ src,
                               _Float16* __restrict__ dst, int n) {
    for (int i = blockIdx.x * blockDim.x + threadIdx.x; i < n; i += gridDim.x * blockDim.x)
        dst[i] = (_Float16)src[i];
}

// ---------------------------------------------------------------- BiLSTM
// grid = 2 blocks (dir 0 = forward, 1 = backward), block = 1024 (32 waves).
// Wave w owns gate columns [16w,16w+16): tiles w, w+32, w+64, w+96 of the
// 16x2048 gate matrix => i/f/g/o for the same column range stay in-register.
// K-loop is split into an x-pass and an h-pass and kept rolled so only one
// A operand and one B operand are live -> no scratch spills around v_wmma.
__global__ __launch_bounds__(1024) void bilstm_wmma(
    const _Float16* __restrict__ emb16,   // (B,L,WORD) f16
    const int*      __restrict__ length,  // (B,)
    const _Float16* __restrict__ WihF, const _Float16* __restrict__ WhhF,
    const float*    __restrict__ bF,
    const _Float16* __restrict__ WihB, const _Float16* __restrict__ WhhB,
    const float*    __restrict__ bB,
    float* __restrict__ hs, float* __restrict__ cs)   // (B,L,1024)
{
    __shared__ _Float16 xh[BATCH * WORDD];   // 16 KB  A-operand x_t
    __shared__ _Float16 hh[BATCH * HALFD];   // 16 KB  A-operand h_{t-1}
    __shared__ int slen[BATCH];

    const int tid  = threadIdx.x;
    const int wave = tid >> 5;
    const int lane = tid & 31;
    const int dir  = blockIdx.x;

    const _Float16* Wih = dir ? WihB : WihF;
    const _Float16* Whh = dir ? WhhB : WhhF;
    const float*    bia = dir ? bB   : bF;

    for (int i = tid; i < BATCH * HALFD; i += 1024) hh[i] = (_Float16)0.0f;
    if (tid < BATCH) slen[tid] = length[tid];
    __syncthreads();

    const int ncol = (wave << 4) + (lane & 15);   // hidden unit [0,512)
    const int arow = lane & 15;
    const int asel = (lane < 16) ? 0 : 16;
    const int bsel = (lane < 16) ? 0 : 16;
    const float bi = bia[ncol], bf = bia[512 + ncol],
                bg = bia[1024 + ncol], bo = bia[1536 + ncol];

    v8f creg = {};

    #pragma unroll 1
    for (int t = 0; t < LSEQ; ++t) {
        // ---- stage x_t into LDS (f16) ----
        if (dir == 0) {
            if (wave == 0) {
                // Tensor Data Mover: 16x512 f16 tile, batch-row stride L*WORD.
                uint64_t ga  = (uint64_t)(uintptr_t)(emb16 + (size_t)t * WORDD);
                uint32_t lds = (uint32_t)(uintptr_t)(&xh[0]);
                v4u g0 = { 1u,                                  // count=1
                           lds,
                           (uint32_t)ga,
                           (uint32_t)((ga >> 32) & 0x1FFFFFFu) | (2u << 30) }; // type=2
                v8i g1 = { (int)(1u << 16),          // data_size=2B
                           (int)(512u << 16),        // tensor_dim0=512 (lo16<<16)
                           (int)(16u << 16),         // tensor_dim1=16
                           (int)(512u << 16),        // tile_dim0=512
                           16,                       // tile_dim1=16, tile_dim2=0
                           (int)(LSEQ * WORDD),      // tensor_dim0_stride=16384
                           0, 0 };
                v4i z4 = { 0, 0, 0, 0 };
                v8i z8 = { 0, 0, 0, 0, 0, 0, 0, 0 };
                __builtin_amdgcn_tensor_load_to_lds(g0, g1, z4, z4, z8, 0);
                __builtin_amdgcn_s_wait_tensorcnt(0);
            }
        } else {
            for (int i = tid; i < BATCH * WORDD; i += 1024) {
                int b = i >> 9, k = i & 511;
                int lb = slen[b];
                int st = (t < lb) ? (lb - 1 - t) : t;
                xh[i] = emb16[(b * LSEQ + st) * WORDD + k];
            }
        }
        __syncthreads();

        v8f ai = {}, af = {}, ag = {}, ao = {};
        const _Float16* bih = Wih + (size_t)ncol * WORDD;
        const _Float16* bhh = Whh + (size_t)ncol * HALFD;

        // ---- x-pass: gates += x_t @ Wih^T ----
        #pragma unroll 1
        for (int k0 = 0; k0 < 512; k0 += 32) {
            v16h Ax = pack_a(&xh[arow * WORDD + k0 + asel]);
            v16h B;
            B = *(const v16h*)(bih + 0 * GSTRIDE + k0 + bsel);  ai = wmma_f16(Ax, B, ai);
            B = *(const v16h*)(bih + 1 * GSTRIDE + k0 + bsel);  af = wmma_f16(Ax, B, af);
            B = *(const v16h*)(bih + 2 * GSTRIDE + k0 + bsel);  ag = wmma_f16(Ax, B, ag);
            B = *(const v16h*)(bih + 3 * GSTRIDE + k0 + bsel);  ao = wmma_f16(Ax, B, ao);
        }
        // ---- h-pass: gates += h_{t-1} @ Whh^T ----
        #pragma unroll 1
        for (int k0 = 0; k0 < 512; k0 += 32) {
            v16h Ah = pack_a(&hh[arow * HALFD + k0 + asel]);
            v16h B;
            B = *(const v16h*)(bhh + 0 * GSTRIDE + k0 + bsel);  ai = wmma_f16(Ah, B, ai);
            B = *(const v16h*)(bhh + 1 * GSTRIDE + k0 + bsel);  af = wmma_f16(Ah, B, af);
            B = *(const v16h*)(bhh + 2 * GSTRIDE + k0 + bsel);  ag = wmma_f16(Ah, B, ag);
            B = *(const v16h*)(bhh + 3 * GSTRIDE + k0 + bsel);  ao = wmma_f16(Ah, B, ao);
        }
        __syncthreads();    // all reads of hh done before overwrite

        // ---- elementwise LSTM cell; C/D layout: vgpr r -> row r (lanes<16) / r+8 ----
        #pragma unroll
        for (int r = 0; r < 8; ++r) {
            int brow = (lane & 16) ? (r + 8) : r;
            float iv = sigf(ai[r] + bi);
            float fv = sigf(af[r] + bf);
            float gv = tanhf(ag[r] + bg);
            float ov = sigf(ao[r] + bo);
            float cv = fv * creg[r] + iv * gv;
            creg[r] = cv;
            float hv = ov * tanhf(cv);
            int lb = slen[brow];
            int ts = dir ? ((t < lb) ? (lb - 1 - t) : t) : t;
            size_t o = ((size_t)(brow * LSEQ + ts)) * HDIM + dir * HALFD + ncol;
            hs[o] = hv;
            cs[o] = cv;
            hh[brow * HALFD + ncol] = (_Float16)hv;
        }
        __syncthreads();
    }
}

// ---------------------------------------------------------------- tree build
// One block (256 threads = 8 waves) per sentence.  Straight-through gate's
// forward value is exactly one-hot(argmax), so only argmax of rank logits is
// needed.  Composition matvec (5120x3072) runs on WMMA with v in B column 0.
__global__ __launch_bounds__(256) void tree_wmma(
    const float* __restrict__ emb,        // (B,L,WORD) f32
    const int*   __restrict__ length,
    const float* __restrict__ W1,         // (256,513)
    const float* __restrict__ W2,         // (1,256)
    const _Float16* __restrict__ Wc16,    // (5120,3072) f16
    const float* __restrict__ bc,         // (5120,)
    const float* __restrict__ hs, const float* __restrict__ cs,  // (B,L,1024)
    float* __restrict__ nodeH, float* __restrict__ nodeC,        // (B,64,1024)
    float* __restrict__ out)              // h (16,1024) then c (16,1024)
{
    __shared__ int s_cnt;
    __shared__ int s_s[MAXN], s_e[MAXN], s_k[MAXN], s_l[MAXN], s_r[MAXN];
    __shared__ float xrow[WORDD];
    __shared__ float red[RHID];
    __shared__ float scores[LSEQ];
    __shared__ _Float16 vbuf[CDIM];       // 6 KB
    __shared__ float gbuf[GCOMP];         // 20 KB

    const int tid  = threadIdx.x;
    const int wave = tid >> 5;
    const int lane = tid & 31;
    const int sent = blockIdx.x;
    const int len  = length[sent];
    float* nH = nodeH + (size_t)sent * MAXN * HDIM;
    float* nC = nodeC + (size_t)sent * MAXN * HDIM;

    if (tid == 0) { s_cnt = 1; s_s[0] = 0; s_e[0] = len; s_k[0] = 0; s_l[0] = -1; s_r[0] = -1; }
    __syncthreads();

    // ---- phase A: discover tree structure (splits from emb only) ----
    for (int node = 0; ; ++node) {
        __syncthreads();
        int total = s_cnt;
        if (node >= total) break;
        int s = s_s[node], e = s_e[node], l = e - s;
        if (l >= 2) {
            for (int p = 0; p < l; ++p) {
                for (int i = tid; i < WORDD; i += RHID)
                    xrow[i] = emb[(size_t)(sent * LSEQ + s + p) * WORDD + i];
                __syncthreads();
                float pv = 2.0f * (float)p / (float)l - 1.0f;
                const float* w1r = W1 + (size_t)tid * 513;
                float hj = w1r[512] * pv;
                for (int k = 0; k < WORDD; ++k) hj += w1r[k] * xrow[k];
                red[tid] = fmaxf(hj, 0.0f) * W2[tid];
                __syncthreads();
                for (int off = RHID / 2; off > 0; off >>= 1) {
                    if (tid < off) red[tid] += red[tid + off];
                    __syncthreads();
                }
                if (tid == 0) scores[p] = red[0];
                __syncthreads();
            }
            if (tid == 0) {
                int k = 0; float best = scores[0];
                for (int i = 1; i < l; ++i)
                    if (scores[i] > best) { best = scores[i]; k = i; }
                s_k[node] = k;
                int lc = -1, rc = -1, c = s_cnt;
                if (k > 0)         { lc = c; s_s[c] = s;         s_e[c] = s + k; s_l[c] = -1; s_r[c] = -1; ++c; }
                if (s + k + 1 < e) { rc = c; s_s[c] = s + k + 1; s_e[c] = e;     s_l[c] = -1; s_r[c] = -1; ++c; }
                s_l[node] = lc; s_r[node] = rc; s_cnt = c;
            }
        }
        __syncthreads();
    }
    __syncthreads();
    const int total = s_cnt;

    // ---- leaves: (h,c) = lstm state at that token ----
    for (int node = 0; node < total; ++node) {
        if (s_e[node] - s_s[node] == 1) {
            size_t src = (size_t)(sent * LSEQ + s_s[node]) * HDIM;
            for (int j = tid; j < HDIM; j += RHID) {
                nH[node * HDIM + j] = hs[src + j];
                nC[node * HDIM + j] = cs[src + j];
            }
        }
    }
    __syncthreads();

    // ---- phase B: bottom-up compose (reverse creation order) ----
    for (int node = total - 1; node >= 0; --node) {
        int s = s_s[node], e = s_e[node];
        if (e - s < 2) continue;
        int lc = s_l[node], rc = s_r[node];
        int xi = s + s_k[node];
        size_t xs = (size_t)(sent * LSEQ + xi) * HDIM;
        for (int j = tid; j < HDIM; j += RHID) {
            vbuf[j]          = (_Float16)(lc >= 0 ? nH[lc * HDIM + j] : 0.0f);
            vbuf[HDIM + j]   = (_Float16)(rc >= 0 ? nH[rc * HDIM + j] : 0.0f);
            vbuf[2*HDIM + j] = (_Float16)hs[xs + j];
        }
        __syncthreads();

        // g = Wc (5120x3072) * v via WMMA, v in B-matrix column 0
        const int asel = (lane < 16) ? 0 : 16;
        for (int tile = wave; tile < GCOMP / 16; tile += 8) {
            v8f acc = {};
            const _Float16* Ar = Wc16 + (size_t)(tile * 16 + (lane & 15)) * CDIM;
            #pragma unroll 1
            for (int k0 = 0; k0 < CDIM; k0 += 32) {
                v16h A = pack_a(Ar + k0 + asel);
                v16h B = {};
                if (lane == 0)       B = *(const v16h*)&vbuf[k0];
                else if (lane == 16) B = *(const v16h*)&vbuf[k0 + 16];
                acc = wmma_f16(A, B, acc);
            }
            if ((lane & 15) == 0) {
                int r0 = tile * 16 + ((lane == 16) ? 8 : 0);
                #pragma unroll
                for (int r = 0; r < 8; ++r) gbuf[r0 + r] = acc[r];
            }
        }
        __syncthreads();

        for (int j = tid; j < HDIM; j += RHID) {
            float iv = sigf(gbuf[j]          + bc[j]);
            float fl = sigf(gbuf[HDIM + j]   + bc[HDIM + j]);
            float fr = sigf(gbuf[2*HDIM + j] + bc[2*HDIM + j]);
            float uv = tanhf(gbuf[3*HDIM + j] + bc[3*HDIM + j]);
            float ov = sigf(gbuf[4*HDIM + j] + bc[4*HDIM + j]);
            float cl = (lc >= 0) ? nC[lc * HDIM + j] : 0.0f;
            float cr = (rc >= 0) ? nC[rc * HDIM + j] : 0.0f;
            float cv = fl * cl + fr * cr + iv * uv;
            nC[node * HDIM + j] = cv;
            nH[node * HDIM + j] = ov * tanhf(cv);
        }
        __syncthreads();
    }

    for (int j = tid; j < HDIM; j += RHID) {
        out[(size_t)sent * HDIM + j]           = nH[j];
        out[(size_t)(BATCH + sent) * HDIM + j] = nC[j];
    }
}

// ---------------------------------------------------------------- launcher
extern "C" void kernel_launch(void* const* d_in, const int* in_sizes, int n_in,
                              void* d_out, int out_size, void* d_ws, size_t ws_size,
                              hipStream_t stream) {
    const float* emb   = (const float*)d_in[0];
    const int*   len   = (const int*)  d_in[1];
    const float* WihF  = (const float*)d_in[2];
    const float* WhhF  = (const float*)d_in[3];
    const float* bF    = (const float*)d_in[4];
    const float* WihB  = (const float*)d_in[5];
    const float* WhhB  = (const float*)d_in[6];
    const float* bB    = (const float*)d_in[7];
    const float* W1    = (const float*)d_in[8];
    const float* W2    = (const float*)d_in[9];
    const float* Wc    = (const float*)d_in[10];
    const float* bc    = (const float*)d_in[11];
    float* out = (float*)d_out;

    char* ws = (char*)d_ws;
    const size_t NW = 2048u * 512u;          // per LSTM weight matrix
    const size_t NC = 5120u * 3072u;         // W_comp
    const size_t NE = (size_t)BATCH * LSEQ * WORDD;
    _Float16* wihf16f = (_Float16*)(ws);
    _Float16* whhf16f = (_Float16*)(ws + 2097152);
    _Float16* wihf16b = (_Float16*)(ws + 4194304);
    _Float16* whhf16b = (_Float16*)(ws + 6291456);
    _Float16* wc16    = (_Float16*)(ws + 8388608);
    _Float16* emb16   = (_Float16*)(ws + 39845888);
    float*    hsb     = (float*)   (ws + 40370176);
    float*    csb     = (float*)   (ws + 42467328);
    float*    nodeH   = (float*)   (ws + 44564480);
    float*    nodeC   = (float*)   (ws + 48758784);

    cvt_f32_to_f16<<<1024, 256, 0, stream>>>(WihF, wihf16f, (int)NW);
    cvt_f32_to_f16<<<1024, 256, 0, stream>>>(WhhF, whhf16f, (int)NW);
    cvt_f32_to_f16<<<1024, 256, 0, stream>>>(WihB, wihf16b, (int)NW);
    cvt_f32_to_f16<<<1024, 256, 0, stream>>>(WhhB, whhf16b, (int)NW);
    cvt_f32_to_f16<<<4096, 256, 0, stream>>>(Wc,   wc16,    (int)NC);
    cvt_f32_to_f16<<<512,  256, 0, stream>>>(emb,  emb16,   (int)NE);

    bilstm_wmma<<<dim3(2), dim3(1024), 0, stream>>>(
        emb16, len, wihf16f, whhf16f, bF, wihf16b, whhf16b, bB, hsb, csb);

    tree_wmma<<<dim3(BATCH), dim3(256), 0, stream>>>(
        emb, len, W1, W2, wc16, bc, hsb, csb, nodeH, nodeC, out);
}
